// Quantize_67834713473283
// MI455X (gfx1250) — matrix operand order repaired and checked by reading, not compiled, not run on hardware
//
#include <hip/hip_runtime.h>
#include <hip/hip_bf16.h>

#define DIM    512
#define NEMB   4096
#define NROWS  65536           // B*T = 16*4096
#define QELEMS 33554432        // NROWS*DIM
#define NTILES (NEMB / 16)     // 256

typedef __attribute__((ext_vector_type(16))) __bf16 v16bf;
typedef __attribute__((ext_vector_type(8)))  float  v8f;
typedef int v4i __attribute__((vector_size(16)));   // b128 payload type

#if __has_builtin(__builtin_amdgcn_global_load_async_to_lds_b128)
#define USE_ASYNC_LDS 1
#else
#define USE_ASYNC_LDS 0
#endif

#define GPTR(p) ((__attribute__((address_space(1))) v4i*)(p))   // global (AS1)
#define LPTR(p) ((__attribute__((address_space(3))) v4i*)(p))   // LDS    (AS3)

__device__ __forceinline__ void wait_async0() {
#if __has_builtin(__builtin_amdgcn_s_wait_asynccnt)
    __builtin_amdgcn_s_wait_asynccnt(0);
#else
    asm volatile("s_wait_asynccnt 0x0" ::: "memory");
#endif
}

__device__ __forceinline__ __bf16 f2bf(float f) {
    union { float f; unsigned u; } in; in.f = f;
    unsigned r = in.u + 0x7FFFu + ((in.u >> 16) & 1u);   // round-to-nearest-even
    union { unsigned short s; __bf16 h; } out;
    out.s = (unsigned short)(r >> 16);
    return out.h;
}

// ---------------------------------------------------------------------------
// Kernel 1: transpose codebook -> ws (bf16 for WMMA B, f32 for gather),
// compute per-code squared norms, zero the diff accumulator.
// embed is [DIM][NEMB] row-major; embedT is [NEMB][DIM].
// ---------------------------------------------------------------------------
__global__ void vq_prep_kernel(const float* __restrict__ embed,
                               __bf16* __restrict__ ebf,
                               float*  __restrict__ e32,
                               float*  __restrict__ cnorm,
                               float*  __restrict__ diffacc) {
    const int e = blockIdx.x;
    const int t = threadIdx.x;
    float s = 0.f;
    for (int d = t; d < DIM; d += 128) {
        float v = embed[(size_t)d * NEMB + e];
        e32[(size_t)e * DIM + d] = v;
        ebf[(size_t)e * DIM + d] = f2bf(v);
        s += v * v;
    }
    __shared__ float red[128];
    red[t] = s; __syncthreads();
    for (int o = 64; o; o >>= 1) { if (t < o) red[t] += red[t + o]; __syncthreads(); }
    if (t == 0) cnorm[e] = red[0];
    if (e == 0 && t == 0) *diffacc = 0.f;
}

// ---------------------------------------------------------------------------
// Kernel 2: fused distance GEMM + argmin, double-buffered async B staging.
// Block = 256 threads = 8 waves; each wave owns a 16-row M tile (block = 128).
// A (16x512 bf16) lives in registers with k contiguous per lane half (k = 16h + s).
// B tiles (16 codes x 512 K bf16 = 16KB) stream via GLOBAL_LOAD_ASYNC_TO_LDS,
// double-buffered so the copy of tile nt+1 overlaps the 16 WMMAs of tile nt.
// dist = ||e||^2 - 2*(f.e)   (||f||^2 is row-constant, irrelevant to argmin)
// ---------------------------------------------------------------------------
__global__ void __launch_bounds__(256, 1)
vq_gemm_argmin_kernel(const float*  __restrict__ x,
                      const __bf16* __restrict__ ebf,
                      const float*  __restrict__ cnorm,
                      int*          __restrict__ idx_out) {
    __shared__ __bf16 lb[2][16 * DIM];   // 2 x 16 KB B tiles

    const int tid  = threadIdx.x;
    const int wave = tid >> 5;
    const int lane = tid & 31;
    const int m    = lane & 15;       // M row within tile (A) / N col within tile (B,C)
    const int hi   = lane >> 4;       // half-wave selector per ISA layouts
    const int rowbase = blockIdx.x * 128 + wave * 16;

    const int ccol = tid >> 4;        // cooperative B copy: 16 threads per code row
    const int cseg = tid & 15;        // each thread moves 4 x 16B = 64B

    // ---- Kick off tile 0 copy before the (long) A load/convert.
#if USE_ASYNC_LDS
    {
        const __bf16* src = ebf + (size_t)ccol * DIM + cseg * 32;
        __bf16*       dst = &lb[0][ccol * DIM + cseg * 32];
        #pragma unroll
        for (int i = 0; i < 4; ++i)
            __builtin_amdgcn_global_load_async_to_lds_b128(
                GPTR(src + i * 8), LPTR(dst + i * 8), 0, 0);
    }
#endif

    // ---- Load A: this wave's 16 rows x K=512, contiguous k order (matches B):
    // lane<16 holds M=lane, K in {32t+0..15}; lane>=16 holds K+16.
    const float* xrow = x + (size_t)(rowbase + m) * DIM;
    v16bf a[16];
    #pragma unroll
    for (int t = 0; t < 16; ++t) {
        const int k0 = 32 * t + 16 * hi;
        float f[16];
        *(float4*)&f[0]  = *(const float4*)(xrow + k0);
        *(float4*)&f[4]  = *(const float4*)(xrow + k0 + 4);
        *(float4*)&f[8]  = *(const float4*)(xrow + k0 + 8);
        *(float4*)&f[12] = *(const float4*)(xrow + k0 + 12);
        union { v16bf v; __bf16 e[16]; } au;
        #pragma unroll
        for (int j = 0; j < 16; ++j) au.e[j] = f2bf(f[j]);
        a[t] = au.v;
    }

    float minv[8];
    int   mini[8];
    #pragma unroll
    for (int j = 0; j < 8; ++j) { minv[j] = 3.4e38f; mini[j] = 0; }

    for (int nt = 0; nt < NTILES; ++nt) {
        const int buf = nt & 1;

#if USE_ASYNC_LDS
        wait_async0();                // this wave's copies for tile nt landed
        __syncthreads();              // everyone's copies landed; buf^1 consumed
        if (nt + 1 < NTILES) {        // prefetch tile nt+1 into the other buffer
            const __bf16* src = ebf + (size_t)((nt + 1) * 16 + ccol) * DIM + cseg * 32;
            __bf16*       dst = &lb[buf ^ 1][ccol * DIM + cseg * 32];
            #pragma unroll
            for (int i = 0; i < 4; ++i)
                __builtin_amdgcn_global_load_async_to_lds_b128(
                    GPTR(src + i * 8), LPTR(dst + i * 8), 0, 0);
        }
#else
        __syncthreads();              // previous tile consumed
        {
            const uint4* src = (const uint4*)(ebf + (size_t)(nt * 16 + ccol) * DIM) + cseg * 4;
            uint4*       dst = (uint4*)(&lb[buf][ccol * DIM]) + cseg * 4;
            dst[0] = src[0]; dst[1] = src[1]; dst[2] = src[2]; dst[3] = src[3];
        }
        __syncthreads();
#endif

        const __bf16* bt = &lb[buf][0];
        v8f acc = {};
        #pragma unroll
        for (int t = 0; t < 16; ++t) {
            // B layout, contiguous k as in A: lane n<16 -> col n, K = 32t+0..15; lane>=16 -> K+16
            const v16bf b = *(const v16bf*)(bt + m * DIM + 32 * t + 16 * hi);
            acc = __builtin_amdgcn_wmma_f32_16x16x32_bf16(
                      false, a[t], false, b, (short)0, acc, false, false);
        }

        const int   col = nt * 16 + m;
        const float cn  = cnorm[col];
        #pragma unroll
        for (int j = 0; j < 8; ++j) {
            float d = cn - 2.0f * acc[j];       // C layout: VGPR j -> M = j + 8*hi
            if (d < minv[j]) { minv[j] = d; mini[j] = col; }
        }
    }

    // Cross-lane (over the 16 N columns) min+argmin reduce, tie -> lowest index.
    #pragma unroll
    for (int j = 0; j < 8; ++j) {
        float v = minv[j]; int i = mini[j];
        for (int off = 8; off; off >>= 1) {
            float ov = __shfl_xor(v, off, 16);
            int   oi = __shfl_xor(i, off, 16);
            if (ov < v || (ov == v && oi < i)) { v = ov; i = oi; }
        }
        if (m == 0) idx_out[rowbase + j + 8 * hi] = i;
    }
}

// ---------------------------------------------------------------------------
// Kernel 3: gather quantize_st (== quantize) from f32 transposed codebook,
// accumulate sum((q - x)^2), emit embed_ind as float.
// ---------------------------------------------------------------------------
__global__ void vq_gather_kernel(const float* __restrict__ x,
                                 const float* __restrict__ e32,
                                 const int*   __restrict__ idx,
                                 float*       __restrict__ outq,
                                 float*       __restrict__ outind,
                                 float*       __restrict__ diffacc) {
    const int t    = threadIdx.x;
    const int row0 = blockIdx.x * 8;
    const int r    = t >> 5;
    const int lane = t & 31;
    const int row  = row0 + r;
    const int code = idx[row];

    const float4* src = (const float4*)(e32 + (size_t)code * DIM) + lane * 4;
    const float4* xs  = (const float4*)(x   + (size_t)row  * DIM) + lane * 4;
    float4*       dst = (float4*)(outq + (size_t)row * DIM) + lane * 4;

    float s = 0.f;
    #pragma unroll
    for (int i = 0; i < 4; ++i) {
        float4 q  = src[i];
        float4 xv = xs[i];
        dst[i] = q;
        float dx = q.x - xv.x, dy = q.y - xv.y, dz = q.z - xv.z, dw = q.w - xv.w;
        s += dx * dx + dy * dy + dz * dz + dw * dw;
    }

    __shared__ float red[256];
    red[t] = s; __syncthreads();
    for (int o = 128; o; o >>= 1) { if (t < o) red[t] += red[t + o]; __syncthreads(); }
    if (t == 0) atomicAdd(diffacc, red[0]);
    if (t < 8)  outind[row0 + t] = (float)idx[row0 + t];
}

__global__ void vq_finalize_kernel(const float* __restrict__ diffacc,
                                   float*       __restrict__ outdiff) {
    *outdiff = *diffacc * (1.0f / (float)QELEMS);
}

// ---------------------------------------------------------------------------
// Workspace layout (bytes):
//   [0,        4194304)  embedT bf16  [4096][512]
//   [4194304, 12582912)  embedT f32   [4096][512]
//   [12582912,12599296)  code norms f32 [4096]
//   [12599296,12861440)  argmin idx  int [65536]
//   [12861440,12861444)  diff accumulator f32
// d_out: [0,33554432) quantize_st | [33554432] diff | [33554433,..) embed_ind
// ---------------------------------------------------------------------------
extern "C" void kernel_launch(void* const* d_in, const int* in_sizes, int n_in,
                              void* d_out, int out_size, void* d_ws, size_t ws_size,
                              hipStream_t stream) {
    const float* x     = (const float*)d_in[0];
    const float* embed = (const float*)d_in[1];

    char*   ws      = (char*)d_ws;
    __bf16* ebf     = (__bf16*)ws;
    float*  e32     = (float*)(ws + 4194304);
    float*  cnorm   = (float*)(ws + 12582912);
    int*    idx     = (int*)  (ws + 12599296);
    float*  diffacc = (float*)(ws + 12861440);

    float* outq    = (float*)d_out;
    float* outdiff = outq + QELEMS;
    float* outind  = outq + QELEMS + 1;

    vq_prep_kernel<<<NEMB, 128, 0, stream>>>(embed, ebf, e32, cnorm, diffacc);
    vq_gemm_argmin_kernel<<<NROWS / 128, 256, 0, stream>>>(x, ebf, cnorm, idx);
    vq_gather_kernel<<<NROWS / 8, 256, 0, stream>>>(x, e32, idx, outq, outind, diffacc);
    vq_finalize_kernel<<<1, 1, 0, stream>>>(diffacc, outdiff);
}
